// TinyChLSTMEncoder_26525718020278
// MI455X (gfx1250) — matrix operand order repaired and checked by reading, not compiled, not run on hardware
//
#include <hip/hip_runtime.h>

// ---------------------------------------------------------------------------
// TinyChLSTMEncoder for MI455X (gfx1250, wave32, WMMA)
//
// Pipeline:
//   K1: fused depthwise-conv1(k=5,p=2)+BN+GELU -> conv2+BN+GELU, per (b,c)
//       row of T=1000, output written as f16 sequence buffer in d_ws.
//   K2: bidirectional LSTM (input=1, hidden=16) over 4096 independent
//       sequences. One wave32 handles 16 sequences; the per-step gate GEMM
//       gates^T(16x16) = W_hh_tile(16x16+x-slot) x H^T  is done with
//       v_wmma_f32_16x16x32_f16 (4 tiles: i,f,g,o). h/c live in VGPRs.
//   K3: linear (4096 x 32) @ (32 x 32)^T + b -> d_out.
//
// Workspace layout: [0, 8.192MB) f16 seq buffer, [8.192MB, +512KB) f32 feat.
// ---------------------------------------------------------------------------

typedef __attribute__((ext_vector_type(16))) _Float16 v16h;
typedef __attribute__((ext_vector_type(8)))  float    v8f;

#define B_DIM 64
#define C_DIM 64
#define T_DIM 1000
#define H_DIM 16
#define E_DIM 32
#define NSEQ  (B_DIM * C_DIM)   // 4096
#define LOG2E 1.442695040888963f

static __device__ __forceinline__ float fast_sigmoid(float x) {
  return __builtin_amdgcn_rcpf(1.0f + __builtin_amdgcn_exp2f(-LOG2E * x));
}
static __device__ __forceinline__ float fast_tanh(float x) {
  return 2.0f * __builtin_amdgcn_rcpf(1.0f + __builtin_amdgcn_exp2f(-2.0f * LOG2E * x)) - 1.0f;
}
static __device__ __forceinline__ unsigned short h_bits(float a) {
  union { _Float16 h; unsigned short u; } cv; cv.h = (_Float16)a; return cv.u;
}
static __device__ __forceinline__ unsigned int pk16(float a, float b) {
  return (unsigned int)h_bits(a) | ((unsigned int)h_bits(b) << 16);
}

// ---------------------------------------------------------------------------
// Kernel 1: fused conv1+bn1+gelu + conv2+bn2+gelu, one block per (b,c) row.
// ---------------------------------------------------------------------------
__global__ void __launch_bounds__(256)
k_conv_fused(const float* __restrict__ x,
             const float* __restrict__ w1, const float* __restrict__ b1,
             const float* __restrict__ g1, const float* __restrict__ be1,
             const float* __restrict__ m1, const float* __restrict__ v1,
             const float* __restrict__ w2, const float* __restrict__ b2,
             const float* __restrict__ g2, const float* __restrict__ be2,
             const float* __restrict__ m2, const float* __restrict__ v2,
             _Float16* __restrict__ hseq)
{
  __shared__ float s0[T_DIM + 4];
  __shared__ float s1[T_DIM + 4];
  const int row = blockIdx.x;          // b*C + c
  const int ch  = row & (C_DIM - 1);
  const float* xr = x + (long)row * T_DIM;

  if (threadIdx.x < 4) {               // zero halos (pad=2 on each side)
    int i = (threadIdx.x & 1) + (threadIdx.x >> 1) * (T_DIM + 2);
    s0[i] = 0.0f; s1[i] = 0.0f;
  }
  for (int t = threadIdx.x; t < T_DIM; t += blockDim.x) s0[2 + t] = xr[t];
  __syncthreads();

  // stage 1
  {
    const float k0 = w1[ch*5+0], k1 = w1[ch*5+1], k2 = w1[ch*5+2],
                k3 = w1[ch*5+3], k4 = w1[ch*5+4], bb = b1[ch];
    const float sc = g1[ch] * __builtin_amdgcn_rsqf(v1[ch] + 1e-5f);
    const float sh = be1[ch] - m1[ch] * sc;
    for (int t = threadIdx.x; t < T_DIM; t += blockDim.x) {
      float y = s0[t]*k0 + s0[t+1]*k1 + s0[t+2]*k2 + s0[t+3]*k3 + s0[t+4]*k4 + bb;
      y = y * sc + sh;
      y = 0.5f * y * (1.0f + erff(y * 0.70710678118f));   // exact GELU
      s1[2 + t] = y;
    }
  }
  __syncthreads();

  // stage 2, emit f16
  {
    const float k0 = w2[ch*5+0], k1 = w2[ch*5+1], k2 = w2[ch*5+2],
                k3 = w2[ch*5+3], k4 = w2[ch*5+4], bb = b2[ch];
    const float sc = g2[ch] * __builtin_amdgcn_rsqf(v2[ch] + 1e-5f);
    const float sh = be2[ch] - m2[ch] * sc;
    _Float16* out = hseq + (long)row * T_DIM;
    for (int t = threadIdx.x; t < T_DIM; t += blockDim.x) {
      float y = s1[t]*k0 + s1[t+1]*k1 + s1[t+2]*k2 + s1[t+3]*k3 + s1[t+4]*k4 + bb;
      y = y * sc + sh;
      y = 0.5f * y * (1.0f + erff(y * 0.70710678118f));
      out[t] = (_Float16)y;
    }
  }
}

// ---------------------------------------------------------------------------
// Kernel 2: LSTM. grid = (32, 2 dirs), block = 256 (8 waves), 16 seqs/wave.
//
// WMMA mapping per gate tile gt (i,f,g,o = rows gt*16..gt*16+15 of W):
//   A (16x32 f16, constant): M = gate-in-tile, K = [hid 0..15 | w_ih slot | 0]
//   B (32x16 f16, per step): K = [h 0..15 | x slot | 0], N = seq-in-wave
//   C (16x16 f32): bias broadcast.  D lane layout == h/c state layout.
// ---------------------------------------------------------------------------
__global__ void __launch_bounds__(256)
k_lstm(const _Float16* __restrict__ hseq,
       const float* __restrict__ w_ih_f, const float* __restrict__ w_hh_f,
       const float* __restrict__ b_ih_f, const float* __restrict__ b_hh_f,
       const float* __restrict__ w_ih_r, const float* __restrict__ w_hh_r,
       const float* __restrict__ b_ih_r, const float* __restrict__ b_hh_r,
       float* __restrict__ feat)
{
  const int dir  = blockIdx.y;                       // 0 = fwd, 1 = bwd
  const int lane = threadIdx.x & 31;
  const int wv   = threadIdx.x >> 5;
  const int seqbase = (blockIdx.x * 8 + wv) * 16;
  const int ms = lane & 15;                          // seq-in-wave / gate row
  const int hs = lane >> 4;                          // half selector

  const float* w_ih = dir ? w_ih_r : w_ih_f;
  const float* w_hh = dir ? w_hh_r : w_hh_f;
  const float* b_ih = dir ? b_ih_r : b_ih_f;
  const float* b_hh = dir ? b_hh_r : b_hh_f;

  // --- constant A tiles (W_hh rows in f16 A-layout, w_ih in K=16 slot) ---
  union AV { v16h v; unsigned int u[8]; } A0, A1, A2, A3;
  {
    union AV* Ap[4] = { &A0, &A1, &A2, &A3 };
#pragma unroll
    for (int gt = 0; gt < 4; ++gt) {
      const float* wr = w_hh + (gt * 16 + ms) * H_DIM + 8 * hs;
#pragma unroll
      for (int j = 0; j < 4; ++j) {
        float2 w2 = *(const float2*)(wr + 2 * j);
        Ap[gt]->u[j] = pk16(w2.x, w2.y);
      }
      unsigned int wi = pk16(w_ih[gt * 16 + ms], 0.0f);
      Ap[gt]->u[4] = (hs == 0) ? wi : 0u;            // K=16 slot (lanes 0-15)
      Ap[gt]->u[5] = 0u; Ap[gt]->u[6] = 0u; Ap[gt]->u[7] = 0u;
    }
  }
  // --- bias accumulators (C operand; M = v + 8*hs) ---
  v8f cb0, cb1, cb2, cb3;
#pragma unroll
  for (int v = 0; v < 8; ++v) {
    int r = v + 8 * hs;
    cb0[v] = b_ih[ 0 + r] + b_hh[ 0 + r];
    cb1[v] = b_ih[16 + r] + b_hh[16 + r];
    cb2[v] = b_ih[32 + r] + b_hh[32 + r];
    cb3[v] = b_ih[48 + r] + b_hh[48 + r];
  }

  // state: lane holds c/h[seq = ms][hid = v + 8*hs], v = 0..7
  float cst[8], hst[8];
#pragma unroll
  for (int v = 0; v < 8; ++v) { cst[v] = 0.0f; hst[v] = 0.0f; }

  const _Float16* xrow = hseq + (long)(seqbase + ms) * T_DIM;
  const bool lo = (lane < 16);

  for (int m = 0; m < T_DIM / 8; ++m) {
    const int t0 = dir ? (T_DIM - 8 - 8 * m) : (8 * m);
    union { uint4 v4; unsigned short s[8]; } xc;
    xc.v4 = *(const uint4*)(xrow + t0);              // 16B-aligned chunk

#pragma unroll
    for (int r = 0; r < 8; ++r) {
      const unsigned short xf = xc.s[r];
      const unsigned short xb = xc.s[7 - r];
      const unsigned int xu = (unsigned int)(dir ? xb : xf);

      // --- build B = H^T (lanes 0-15: hid 0..7 + x slot; lanes 16-31: hid 8..15) ---
      unsigned int P0 = pk16(hst[0], hst[1]);
      unsigned int P1 = pk16(hst[2], hst[3]);
      unsigned int P2 = pk16(hst[4], hst[5]);
      unsigned int P3 = pk16(hst[6], hst[7]);
      unsigned int S0 = (unsigned int)__shfl((int)P0, ms + 16, 32);
      unsigned int S1 = (unsigned int)__shfl((int)P1, ms + 16, 32);
      unsigned int S2 = (unsigned int)__shfl((int)P2, ms + 16, 32);
      unsigned int S3 = (unsigned int)__shfl((int)P3, ms + 16, 32);
      union { v16h v; unsigned int u[8]; } Bm;
      Bm.u[0] = P0;             // K0,1   | K8,9 (own hid 8*hs+v)
      Bm.u[1] = P1;
      Bm.u[2] = P2;
      Bm.u[3] = P3;
      Bm.u[4] = lo ? xu : 0u;   // K16 = x_t, K17 = 0 (lanes 0-15)
      Bm.u[5] = 0u;
      Bm.u[6] = 0u;
      Bm.u[7] = 0u;

      // --- four gate tiles: D = W_tile x H^T + bias ---
      v8f di = __builtin_amdgcn_wmma_f32_16x16x32_f16(false, A0.v, false, Bm.v, (short)0, cb0, false, false);
      v8f df = __builtin_amdgcn_wmma_f32_16x16x32_f16(false, A1.v, false, Bm.v, (short)0, cb1, false, false);
      v8f dg = __builtin_amdgcn_wmma_f32_16x16x32_f16(false, A2.v, false, Bm.v, (short)0, cb2, false, false);
      v8f dv = __builtin_amdgcn_wmma_f32_16x16x32_f16(false, A3.v, false, Bm.v, (short)0, cb3, false, false);

      // --- elementwise LSTM cell update (layout matches D exactly) ---
#pragma unroll
      for (int v = 0; v < 8; ++v) {
        float iv = fast_sigmoid(di[v]);
        float fv = fast_sigmoid(df[v]);
        float gv = fast_tanh(dg[v]);
        float ov = fast_sigmoid(dv[v]);
        cst[v] = fv * cst[v] + iv * gv;
        hst[v] = ov * fast_tanh(cst[v]);
      }
    }
  }

  // final hidden state -> feat[seq][dir*16 + hid]; per lane 8 contiguous f32
  float* dst = feat + (long)(seqbase + ms) * 32 + dir * 16 + 8 * hs;
  float4 o0 = { hst[0], hst[1], hst[2], hst[3] };
  float4 o1 = { hst[4], hst[5], hst[6], hst[7] };
  *(float4*)(dst)     = o0;
  *(float4*)(dst + 4) = o1;
}

// ---------------------------------------------------------------------------
// Kernel 3: out[seq][e] = feat[seq][:] . lin_w[e][:] + lin_b[e]
// ---------------------------------------------------------------------------
__global__ void __launch_bounds__(256)
k_linear(const float* __restrict__ feat, const float* __restrict__ W,
         const float* __restrict__ b, float* __restrict__ out)
{
  const int idx = blockIdx.x * blockDim.x + threadIdx.x;   // 4096*32 exact
  const int seq = idx >> 5, e = idx & 31;
  const float* f = feat + seq * 32;
  const float* w = W + e * 32;
  float acc = b[e];
#pragma unroll
  for (int k = 0; k < 32; ++k) acc += f[k] * w[k];
  out[idx] = acc;
}

// ---------------------------------------------------------------------------
extern "C" void kernel_launch(void* const* d_in, const int* in_sizes, int n_in,
                              void* d_out, int out_size, void* d_ws, size_t ws_size,
                              hipStream_t stream) {
  const float* x       = (const float*)d_in[0];
  const float* conv1_w = (const float*)d_in[1];
  const float* conv1_b = (const float*)d_in[2];
  const float* bn1_g   = (const float*)d_in[3];
  const float* bn1_b   = (const float*)d_in[4];
  const float* bn1_m   = (const float*)d_in[5];
  const float* bn1_v   = (const float*)d_in[6];
  const float* conv2_w = (const float*)d_in[7];
  const float* conv2_b = (const float*)d_in[8];
  const float* bn2_g   = (const float*)d_in[9];
  const float* bn2_b   = (const float*)d_in[10];
  const float* bn2_m   = (const float*)d_in[11];
  const float* bn2_v   = (const float*)d_in[12];
  const float* w_ih_f  = (const float*)d_in[13];
  const float* w_hh_f  = (const float*)d_in[14];
  const float* b_ih_f  = (const float*)d_in[15];
  const float* b_hh_f  = (const float*)d_in[16];
  const float* w_ih_r  = (const float*)d_in[17];
  const float* w_hh_r  = (const float*)d_in[18];
  const float* b_ih_r  = (const float*)d_in[19];
  const float* b_hh_r  = (const float*)d_in[20];
  const float* lin_w   = (const float*)d_in[21];
  const float* lin_b   = (const float*)d_in[22];

  _Float16* hseq = (_Float16*)d_ws;                                  // 8,192,000 B
  float*    feat = (float*)((char*)d_ws + (size_t)NSEQ * T_DIM * 2); // 524,288 B

  k_conv_fused<<<NSEQ, 256, 0, stream>>>(x,
      conv1_w, conv1_b, bn1_g, bn1_b, bn1_m, bn1_v,
      conv2_w, conv2_b, bn2_g, bn2_b, bn2_m, bn2_v, hseq);

  k_lstm<<<dim3(32, 2), 256, 0, stream>>>(hseq,
      w_ih_f, w_hh_f, b_ih_f, b_hh_f,
      w_ih_r, w_hh_r, b_ih_r, b_hh_r, feat);

  k_linear<<<(NSEQ * E_DIM) / 256, 256, 0, stream>>>(feat, lin_w, lin_b,
                                                     (float*)d_out);
}
